// RIAttention_38843684225799
// MI455X (gfx1250) — matrix operand order repaired and checked by reading, not compiled, not run on hardware
//
#include <hip/hip_runtime.h>
#include <cstdint>
#include <cstddef>

// Problem constants (match reference)
#define BATCH  2
#define SEQ    512
#define DIM    256
#define NHEAD  8
#define HDIM   32
#define QKVN   768               // 3*DIM
#define SCALE_F 0.17677669529663687f   // HDIM^-0.5

typedef __attribute__((ext_vector_type(4)))  float   v4f;
typedef __attribute__((ext_vector_type(8)))  float   v8f;
typedef __attribute__((ext_vector_type(16))) __bf16  v16bf;

union FragAB { uint4 q[2]; v16bf v; };

// f32 -> bf16, round-to-nearest-even
__device__ __forceinline__ unsigned short f2bf(float f) {
    unsigned u = __builtin_bit_cast(unsigned, f);
    u += 0x7FFFu + ((u >> 16) & 1u);
    return (unsigned short)(u >> 16);
}

// ---------------------------------------------------------------------------
// Kernel B (placed first so the disasm snippet shows the bandwidth-critical
// kernel): fused bias + q.k^T + softmax + p.V for one (b,i) per workgroup.
//   score[h][j] = SCALE * sum_{c in head h} q[c] * (k[j,c] + rioe[b,i,j,c])
// rioe streamed once with NT b128 loads (keep L2 for qkv). 256 threads.
// ---------------------------------------------------------------------------
__global__ void __launch_bounds__(256)
attn_fused_kernel(const float* __restrict__ rioe, const float* __restrict__ qkv,
                  float* __restrict__ attn_out)
{
    __shared__ float sc[NHEAD][SEQ];       // 16 KB scores -> probs

    const int bi = blockIdx.x;             // b*512 + i
    const int b  = bi >> 9;
    const int t  = threadIdx.x;

    const int jphase = t >> 6;             // 4 rows of j handled per sweep
    const int g      = t & 63;             // 64 threads cover one j row (256 ch / 4)
    const int h8     = g >> 3;             // head of this 8-lane subgroup
    const int c4     = g * 4;

    const float* qrow = qkv + (size_t)bi * QKVN;        // q part: cols [0,256)
    const v4f qf = *(const v4f*)(qrow + c4);

    const float* rbase = rioe + (size_t)bi * SEQ * DIM; // rioe[b,i,:,:]
    const float* kbase = qkv + (size_t)b * SEQ * QKVN + DIM;  // k part of row j

    // ---- pass 1: scores (streams 512 KB of rioe per workgroup) ----
#pragma unroll 2
    for (int j = jphase; j < SEQ; j += 4) {
        // branchless speculative prefetch 16 rows ahead (clamped in-bounds)
        const int jp = (j + 16 < SEQ) ? (j + 16) : (SEQ - 1);
        __builtin_prefetch(rbase + (size_t)jp * DIM + c4, 0, 1);
        v4f rv = __builtin_nontemporal_load((const v4f*)(rbase + (size_t)j * DIM + c4));
        v4f kv = *(const v4f*)(kbase + (size_t)j * QKVN + c4);
        float p = qf.x * (rv.x + kv.x) + qf.y * (rv.y + kv.y)
                + qf.z * (rv.z + kv.z) + qf.w * (rv.w + kv.w);
        // 8-lane segmented reduction (one head = 8 consecutive lanes in-wave)
        p += __shfl_xor(p, 1, 32);
        p += __shfl_xor(p, 2, 32);
        p += __shfl_xor(p, 4, 32);
        if ((g & 7) == 0) sc[h8][j] = p * SCALE_F;
    }
    __syncthreads();

    // ---- pass 2: softmax, wave h owns head h ----
    const int wave = t >> 5;
    const int lane = t & 31;
    float mx = -3.0e38f;
    for (int jj = lane; jj < SEQ; jj += 32) mx = fmaxf(mx, sc[wave][jj]);
#pragma unroll
    for (int s = 16; s > 0; s >>= 1) mx = fmaxf(mx, __shfl_xor(mx, s, 32));
    float sum = 0.f;
    for (int jj = lane; jj < SEQ; jj += 32) {
        float e = __expf(sc[wave][jj] - mx);
        sc[wave][jj] = e;
        sum += e;
    }
#pragma unroll
    for (int s = 16; s > 0; s >>= 1) sum += __shfl_xor(sum, s, 32);
    const float inv = 1.0f / sum;
    __syncthreads();

    // ---- pass 3: out[c] = inv * sum_j p[h][j] * v[j,c]; thread t owns c = t,
    //      head = t>>5 == wave, so inv is already resident in this thread.
    //      v loads are wave-coalesced (consecutive c) and L2-resident. ----
    const float* vcol = qkv + (size_t)b * SEQ * QKVN + 2 * DIM + t;  // v part
    float acc = 0.f;
#pragma unroll 4
    for (int j = 0; j < SEQ; ++j)
        acc = fmaf(sc[wave][j], vcol[(size_t)j * QKVN], acc);

    attn_out[(size_t)bi * DIM + t] = acc * inv;
}

// ---------------------------------------------------------------------------
// Kernel A: qkv = x @ w_qkv^T     x:[1024,256] f32, w:[768,256] f32 -> qkv f32
// 128 threads (4 waves). WG tile: 16 (M) x 64 (N); wave w owns 16x16 subtile.
// bf16 WMMA 16x16x32, fp32 accumulate, K = 256 -> 8 WMMA per tile.
// ---------------------------------------------------------------------------
__global__ void __launch_bounds__(128)
qkv_gemm_kernel(const float* __restrict__ x, const float* __restrict__ w,
                float* __restrict__ qkv)
{
    __shared__ unsigned short xs[16 * 256];   // 8 KB  : A tile (M x K) bf16
    __shared__ unsigned short wsd[64 * 256];  // 32 KB : B tile (N x K) bf16

    const int tid  = threadIdx.x;
    const int m0   = blockIdx.x * 16;
    const int nb   = blockIdx.y * 64;

    // stage x rows m0..m0+15 (convert f32->bf16).
    for (int idx = tid; idx < 16 * 64; idx += 128) {
        const int r  = idx >> 6;
        const int c4 = (idx & 63) * 4;
        v4f f = *(const v4f*)(x + (size_t)(m0 + r) * DIM + c4);
        unsigned short* d = &xs[r * 256 + c4];
        d[0] = f2bf(f.x); d[1] = f2bf(f.y); d[2] = f2bf(f.z); d[3] = f2bf(f.w);
    }
    // stage w rows nb..nb+63.
    for (int idx = tid; idx < 64 * 64; idx += 128) {
        const int r  = idx >> 6;
        const int c4 = (idx & 63) * 4;
        v4f f = *(const v4f*)(w + (size_t)(nb + r) * DIM + c4);
        unsigned short* d = &wsd[r * 256 + c4];
        d[0] = f2bf(f.x); d[1] = f2bf(f.y); d[2] = f2bf(f.z); d[3] = f2bf(f.w);
    }
    __syncthreads();

    const int wave = tid >> 5;
    const int lane = tid & 31;
    const int half = lane >> 4;      // 0: lanes 0-15, 1: lanes 16-31
    const int mrow = lane & 15;      // A: M index
    const int ncol = lane & 15;      // B/D: N index

    v8f acc = {};
#pragma unroll
    for (int k0 = 0; k0 < 256; k0 += 32) {
        FragAB A, B;
        // A 16x32 bf16: lanes 0-15 hold K {k0+0..7, k0+16..23}; lanes 16-31 the other halves
        A.q[0] = *(const uint4*)&xs[mrow * 256 + k0 + half * 8];
        A.q[1] = *(const uint4*)&xs[mrow * 256 + k0 + 16 + half * 8];
        // B 32x16 bf16 (K x N): lanes 0-15 hold K {k0+0..7, k0+16..23}; lanes 16-31 the other halves
        B.q[0] = *(const uint4*)&wsd[(wave * 16 + ncol) * 256 + k0 + half * 8];
        B.q[1] = *(const uint4*)&wsd[(wave * 16 + ncol) * 256 + k0 + 16 + half * 8];
        acc = __builtin_amdgcn_wmma_f32_16x16x32_bf16(
            false, A.v, false, B.v, (short)0, acc, false, false);
    }

    // D layout: VGPR r -> M = r + 8*half ; N = lane%16
    const int n = nb + wave * 16 + ncol;
#pragma unroll
    for (int r = 0; r < 8; ++r) {
        const int m = m0 + r + half * 8;
        qkv[(size_t)m * QKVN + n] = acc[r];
    }
}

// ---------------------------------------------------------------------------
// Kernel C: out = attn_out @ w_proj^T + b_proj    (M=1024, N=256, K=256)
// Same WMMA tiling as kernel A.
// ---------------------------------------------------------------------------
__global__ void __launch_bounds__(128)
proj_gemm_kernel(const float* __restrict__ a, const float* __restrict__ w,
                 const float* __restrict__ bias, float* __restrict__ out)
{
    __shared__ unsigned short xs[16 * 256];
    __shared__ unsigned short wsd[64 * 256];

    const int tid = threadIdx.x;
    const int m0  = blockIdx.x * 16;
    const int nb  = blockIdx.y * 64;

    for (int idx = tid; idx < 16 * 64; idx += 128) {
        const int r  = idx >> 6;
        const int c4 = (idx & 63) * 4;
        v4f f = *(const v4f*)(a + (size_t)(m0 + r) * DIM + c4);
        unsigned short* d = &xs[r * 256 + c4];
        d[0] = f2bf(f.x); d[1] = f2bf(f.y); d[2] = f2bf(f.z); d[3] = f2bf(f.w);
    }
    for (int idx = tid; idx < 64 * 64; idx += 128) {
        const int r  = idx >> 6;
        const int c4 = (idx & 63) * 4;
        v4f f = *(const v4f*)(w + (size_t)(nb + r) * DIM + c4);
        unsigned short* d = &wsd[r * 256 + c4];
        d[0] = f2bf(f.x); d[1] = f2bf(f.y); d[2] = f2bf(f.z); d[3] = f2bf(f.w);
    }
    __syncthreads();

    const int wave = tid >> 5;
    const int lane = tid & 31;
    const int half = lane >> 4;
    const int mrow = lane & 15;
    const int ncol = lane & 15;

    v8f acc = {};
#pragma unroll
    for (int k0 = 0; k0 < 256; k0 += 32) {
        FragAB A, B;
        A.q[0] = *(const uint4*)&xs[mrow * 256 + k0 + half * 8];
        A.q[1] = *(const uint4*)&xs[mrow * 256 + k0 + 16 + half * 8];
        B.q[0] = *(const uint4*)&wsd[(wave * 16 + ncol) * 256 + k0 + half * 8];
        B.q[1] = *(const uint4*)&wsd[(wave * 16 + ncol) * 256 + k0 + 16 + half * 8];
        acc = __builtin_amdgcn_wmma_f32_16x16x32_bf16(
            false, A.v, false, B.v, (short)0, acc, false, false);
    }

    const int n  = nb + wave * 16 + ncol;
    const float bn = bias[n];
#pragma unroll
    for (int r = 0; r < 8; ++r) {
        const int m = m0 + r + half * 8;
        out[(size_t)m * DIM + n] = acc[r] + bn;
    }
}

// ---------------------------------------------------------------------------
extern "C" void kernel_launch(void* const* d_in, const int* in_sizes, int n_in,
                              void* d_out, int out_size, void* d_ws, size_t ws_size,
                              hipStream_t stream)
{
    (void)in_sizes; (void)n_in; (void)out_size; (void)ws_size;
    const float* x      = (const float*)d_in[0];   // [2,512,256]
    const float* rioe   = (const float*)d_in[1];   // [2,512,512,256]
    const float* w_qkv  = (const float*)d_in[2];   // [768,256]
    const float* w_proj = (const float*)d_in[3];   // [256,256]
    const float* b_proj = (const float*)d_in[4];   // [256]
    float* out = (float*)d_out;                    // [2,512,256]

    float* qkv  = (float*)d_ws;                       // [1024,768] fp32 (3 MB)
    float* attn = qkv + (size_t)BATCH * SEQ * QKVN;   // [1024,256] fp32 (1 MB)

    // 1) qkv = x @ w_qkv^T
    qkv_gemm_kernel<<<dim3(BATCH * SEQ / 16, QKVN / 64), 128, 0, stream>>>(x, w_qkv, qkv);
    // 2) fused bias + q.k^T + softmax + p.V  (one WG per (b,i))
    attn_fused_kernel<<<dim3(BATCH * SEQ), 256, 0, stream>>>(rioe, qkv, attn);
    // 3) out = attn @ w_proj^T + b_proj
    proj_gemm_kernel<<<dim3(BATCH * SEQ / 16, DIM / 64), 128, 0, stream>>>(attn, w_proj, b_proj, out);
}
